// SlotAttention_46480136077837
// MI455X (gfx1250) — matrix operand order repaired and checked by reading, not compiled, not run on hardware
//
#include <hip/hip_runtime.h>
#include <hip/hip_bf16.h>

// ---------- WMMA types (gfx1250, wave32) ----------
typedef __attribute__((ext_vector_type(16))) _Float16 v16h;
typedef __attribute__((ext_vector_type(8)))  _Float16 v8h;
typedef __attribute__((ext_vector_type(8)))  float    v8f;

union AFrag { v16h v; v8h h[2]; };

#define WMMA_F16(a, b, c) \
  __builtin_amdgcn_wmma_f32_16x16x32_f16(false, (a), false, (b), (short)0, (c), false, false)

static constexpr int B_ = 64, S_ = 8, F_ = 4096, DIN_ = 256, D_ = 64, H_ = 256;

__device__ __forceinline__ float wave_sum32(float s) {
  #pragma unroll
  for (int o = 16; o >= 1; o >>= 1) s += __shfl_xor(s, o, 32);
  return s;
}

// ---------- K0: convert Wk, Wv to f16 ----------
__global__ void k_cvt_w(const float* __restrict__ Wk, const float* __restrict__ Wv,
                        _Float16* __restrict__ Wk16, _Float16* __restrict__ Wv16) {
  int i = blockIdx.x * blockDim.x + threadIdx.x;
  if (i < D_ * DIN_) {
    Wk16[i] = (_Float16)Wk[i];
    Wv16[i] = (_Float16)Wv[i];
  }
}

// ---------- K1: LN(features) fused with keys/values projection (WMMA) ----------
// grid (32, 64): x = 64-row block within batch, y = batch. 128 threads = 4 waves,
// each wave owns 16 rows (one WMMA M-tile).
// keys16:  (B, F, 64) row-major f16   (dots B-frags contiguous)
// valT16:  (B, 64, F) f16 transposed  (updates B-frags contiguous, packed stores)
__global__ __launch_bounds__(128) void k_ln_project(
    const float* __restrict__ features, const float* __restrict__ gf, const float* __restrict__ bf,
    const _Float16* __restrict__ Wk16, const _Float16* __restrict__ Wv16,
    _Float16* __restrict__ keys16, _Float16* __restrict__ valT16) {
  const int b    = blockIdx.y;
  const int w    = threadIdx.x >> 5;
  const int lane = threadIdx.x & 31;
  const int m0   = blockIdx.x * 64 + w * 16;

  __shared__ _Float16 At[4][16][264];   // 264 = 256 + 8 halves pad

  float gfr[8], bfr[8];
  #pragma unroll
  for (int i = 0; i < 8; ++i) { gfr[i] = gf[lane * 8 + i]; bfr[i] = bf[lane * 8 + i]; }

  // LayerNorm 16 rows (one row per pass, 8 elems/lane) into LDS as f16
  for (int r = 0; r < 16; ++r) {
    const float* x = features + ((size_t)b * F_ + (m0 + r)) * DIN_ + lane * 8;
    float v[8], s = 0.f, s2 = 0.f;
    #pragma unroll
    for (int i = 0; i < 8; ++i) { v[i] = x[i]; s += v[i]; s2 += v[i] * v[i]; }
    s = wave_sum32(s); s2 = wave_sum32(s2);
    float mu   = s * (1.f / DIN_);
    float rstd = rsqrtf(s2 * (1.f / DIN_) - mu * mu + 1e-5f);
    #pragma unroll
    for (int i = 0; i < 8; ++i)
      At[w][r][lane * 8 + i] = (_Float16)((v[i] - mu) * rstd * gfr[i] + bfr[i]);
  }
  __syncthreads();

  // A fragments: lane m holds K 0..15 (lanes<16) / 16..31 (lanes>=16)
  const int mrow = lane & 15;
  const int aoff = (lane & 16) ? 16 : 0;
  const int boff = (lane & 16) ? 16 : 0;
  AFrag a[8];
  #pragma unroll
  for (int kc = 0; kc < 8; ++kc) {
    const _Float16* p = &At[w][mrow][kc * 32 + aoff];
    a[kc].h[0] = *(const v8h*)p;
    a[kc].h[1] = *(const v8h*)(p + 8);
  }

  #pragma unroll
  for (int nt = 0; nt < 8; ++nt) {                 // 4 N-tiles keys + 4 N-tiles values
    const _Float16* W = (nt < 4) ? Wk16 : Wv16;
    const int ncol = (nt & 3) * 16 + (lane & 15);  // output feature (column of B)
    v8f acc0 = {}, acc1 = {};
    #pragma unroll
    for (int kc = 0; kc < 8; kc += 2) {            // 2 accumulators: expose WMMA ILP
      v16h b0 = *(const v16h*)(W + ncol * DIN_ + kc * 32 + boff);
      v16h b1 = *(const v16h*)(W + ncol * DIN_ + (kc + 1) * 32 + boff);
      acc0 = WMMA_F16(a[kc].v, b0, acc0);
      acc1 = WMMA_F16(a[kc + 1].v, b1, acc1);
    }
    v8f acc = acc0 + acc1;
    const int rbase = m0 + ((lane & 16) ? 8 : 0);
    if (nt < 4) {
      #pragma unroll
      for (int r = 0; r < 8; ++r)                  // 16-lane row segments, coalesced
        keys16[((size_t)b * F_ + rbase + r) * D_ + ncol] = (_Float16)acc[r];
    } else {
      v8h pk;                                      // 8 contiguous halves -> one b128
      #pragma unroll
      for (int r = 0; r < 8; ++r) pk[r] = (_Float16)acc[r];
      *(v8h*)(valT16 + ((size_t)b * D_ + ncol) * F_ + rbase) = pk;
    }
  }
}

// ---------- K2a: s_n = LN(slots); q = s_n @ Wq^T (f16, rows 8..15 zero-padded) ----------
__global__ __launch_bounds__(256) void k_slots_norm_q(
    const float* __restrict__ slots, const float* __restrict__ gs, const float* __restrict__ bs,
    const float* __restrict__ Wq, float* __restrict__ sn, _Float16* __restrict__ q16) {
  const int b = blockIdx.x, w = threadIdx.x >> 5, lane = threadIdx.x & 31;
  __shared__ float snl[S_][D_];

  {   // wave w handles slot row w
    const float* x = slots + ((size_t)b * S_ + w) * D_ + lane * 2;
    float v0 = x[0], v1 = x[1];
    float s  = wave_sum32(v0 + v1);
    float s2 = wave_sum32(v0 * v0 + v1 * v1);
    float mu = s * (1.f / D_);
    float rstd = rsqrtf(s2 * (1.f / D_) - mu * mu + 1e-5f);
    float y0 = (v0 - mu) * rstd * gs[lane * 2]     + bs[lane * 2];
    float y1 = (v1 - mu) * rstd * gs[lane * 2 + 1] + bs[lane * 2 + 1];
    snl[w][lane * 2] = y0; snl[w][lane * 2 + 1] = y1;
    sn[((size_t)b * S_ + w) * D_ + lane * 2]     = y0;
    sn[((size_t)b * S_ + w) * D_ + lane * 2 + 1] = y1;
  }
  __syncthreads();

  #pragma unroll
  for (int i = 0; i < 2; ++i) {           // 512 q outputs
    int idx = threadIdx.x + i * 256;
    int s = idx >> 6, e = idx & 63;
    float acc = 0.f;
    for (int d = 0; d < D_; ++d) acc += snl[s][d] * Wq[e * D_ + d];
    q16[((size_t)b * 16 + s) * D_ + e] = (_Float16)acc;
  }
  #pragma unroll
  for (int i = 0; i < 2; ++i) {           // zero pad rows 8..15
    int idx = threadIdx.x + i * 256;
    q16[((size_t)b * 16 + 8 + (idx >> 6)) * D_ + (idx & 63)] = (_Float16)0.f;
  }
}

// ---------- K2b: dots (WMMA) + softmax over slots + F-normalizer ----------
// One block per batch, 8 waves stride over 256 F-tiles. Lane L<16 owns column
// f0+L: its 8 C-registers are exactly the 8 slot logits -> in-lane softmax.
// p16 rows 8..15 are pre-zeroed once by hipMemsetAsync (they never change).
__global__ __launch_bounds__(256) void k_attn(
    const _Float16* __restrict__ q16, const _Float16* __restrict__ keys16,
    _Float16* __restrict__ p16, float* __restrict__ attn_out, float* __restrict__ invR) {
  const int b = blockIdx.x, w = threadIdx.x >> 5, lane = threadIdx.x & 31;
  __shared__ float Rs[S_];
  if (threadIdx.x < S_) Rs[threadIdx.x] = 0.f;
  __syncthreads();

  const int aoff = (lane & 16) ? 16 : 0;
  const int boff = (lane & 16) ? 16 : 0;
  AFrag a[2];
  #pragma unroll
  for (int kc = 0; kc < 2; ++kc) {
    const _Float16* p = q16 + (size_t)b * 16 * D_ + (lane & 15) * D_ + kc * 32 + aoff;
    a[kc].h[0] = *(const v8h*)p;
    a[kc].h[1] = *(const v8h*)(p + 8);
  }

  float rsum[S_] = {0.f, 0.f, 0.f, 0.f, 0.f, 0.f, 0.f, 0.f};
  for (int t = w; t < F_ / 16; t += 8) {
    const int f0 = t * 16;
    v8f acc = {};
    #pragma unroll
    for (int kc = 0; kc < 2; ++kc) {
      v16h bf16 = *(const v16h*)(keys16 + ((size_t)b * F_ + f0 + (lane & 15)) * D_ + kc * 32 + boff);
      acc = WMMA_F16(a[kc].v, bf16, acc);
    }
    if (lane < 16) {
      const int f = f0 + lane;
      float d0[S_], m = -3.4e38f;
      #pragma unroll
      for (int r = 0; r < S_; ++r) { d0[r] = acc[r] * 0.125f; m = fmaxf(m, d0[r]); }
      float sum = 0.f;
      #pragma unroll
      for (int r = 0; r < S_; ++r) { d0[r] = __expf(d0[r] - m); sum += d0[r]; }
      const float inv = 1.f / sum;
      #pragma unroll
      for (int r = 0; r < S_; ++r) {
        float pr = d0[r] * inv;
        attn_out[((size_t)b * S_ + r) * F_ + f] = pr;           // pre_norm_attn
        p16[((size_t)b * 16 + r) * F_ + f] = (_Float16)pr;
        rsum[r] += pr;
      }
    }
  }
  #pragma unroll
  for (int r = 0; r < S_; ++r) {
    float s = wave_sum32(rsum[r]);          // lanes>=16 contribute 0
    if (lane == 0) atomicAdd(&Rs[r], s);    // ds_add_f32
  }
  __syncthreads();
  if (threadIdx.x < S_)
    invR[b * S_ + threadIdx.x] = 1.f / (Rs[threadIdx.x] + (float)F_ * 1e-8f);
}

// ---------- K2c: updates (WMMA, K split across 2 wave-groups) + GRU + MLP ----------
__global__ __launch_bounds__(256) void k_update(
    const _Float16* __restrict__ p16, const _Float16* __restrict__ valT16,
    const float* __restrict__ invR, const float* __restrict__ sn,
    const float* __restrict__ W_ih, const float* __restrict__ W_hh,
    const float* __restrict__ b_ih, const float* __restrict__ b_hh,
    const float* __restrict__ gm, const float* __restrict__ bm,
    const float* __restrict__ W1, const float* __restrict__ b1,
    const float* __restrict__ W2, const float* __restrict__ b2,
    float* __restrict__ slots, float* __restrict__ slots_out) {
  const int b = blockIdx.x, w = threadIdx.x >> 5, lane = threadIdx.x & 31;
  const int nt = w & 3, kh = w >> 2;             // N-tile, K-half
  __shared__ float updp[2][S_][D_];              // partial sums per K-half
  __shared__ float snl[S_][D_], ns[S_][D_ + 4], lnn[S_][D_ + 4];
  __shared__ float h1[S_][H_], mus[S_], rstds[S_], ir[S_];

  for (int i = threadIdx.x; i < S_ * D_; i += 256)
    snl[i >> 6][i & 63] = sn[(size_t)b * S_ * D_ + i];
  if (threadIdx.x < S_) ir[threadIdx.x] = invR[b * S_ + threadIdx.x];

  // updates partial: M=16, N=16 (tile nt), K=2048 (half kh)
  const int aoff = (lane & 16) ? 16 : 0;
  const int boff = (lane & 16) ? 16 : 0;
  const int ncol = nt * 16 + (lane & 15);
  const _Float16* arow = p16 + (size_t)b * 16 * F_ + (lane & 15) * F_;
  const _Float16* brow = valT16 + ((size_t)b * D_ + ncol) * F_;
  v8f acc0 = {}, acc1 = {};
  const int kc0 = kh * (F_ / 64);
  for (int kc = kc0; kc < kc0 + F_ / 64; kc += 2) {
    AFrag a0, a1;
    const _Float16* ap = arow + kc * 32 + aoff;
    a0.h[0] = *(const v8h*)ap;        a0.h[1] = *(const v8h*)(ap + 8);
    a1.h[0] = *(const v8h*)(ap + 32); a1.h[1] = *(const v8h*)(ap + 40);
    v16h bf0 = *(const v16h*)(brow + kc * 32 + boff);
    v16h bf1 = *(const v16h*)(brow + (kc + 1) * 32 + boff);
    acc0 = WMMA_F16(a0.v, bf0, acc0);
    acc1 = WMMA_F16(a1.v, bf1, acc1);
  }
  v8f acc = acc0 + acc1;
  if (lane < 16) {
    #pragma unroll
    for (int r = 0; r < S_; ++r) updp[kh][r][ncol] = acc[r];
  }
  __syncthreads();

  // combine K-halves and apply 1/R: 512 elems, 2 per thread
  #pragma unroll
  for (int i = 0; i < 2; ++i) {
    int idx = threadIdx.x + i * 256;
    int s = idx >> 6, d = idx & 63;
    updp[0][s][d] = (updp[0][s][d] + updp[1][s][d]) * ir[s];
  }
  __syncthreads();

  // GRU (r,z,n order): 512 outputs, 2 per thread
  float newh[2];
  #pragma unroll
  for (int i = 0; i < 2; ++i) {
    int idx = threadIdx.x + i * 256;
    int s = idx >> 6, e = idx & 63;
    float gir = b_ih[e], giz = b_ih[D_ + e], gin = b_ih[2 * D_ + e];
    float ghr = b_hh[e], ghz = b_hh[D_ + e], ghn = b_hh[2 * D_ + e];
    for (int d = 0; d < D_; ++d) {
      float u = updp[0][s][d], h = snl[s][d];
      gir += u * W_ih[e * D_ + d];            ghr += h * W_hh[e * D_ + d];
      giz += u * W_ih[(D_ + e) * D_ + d];     ghz += h * W_hh[(D_ + e) * D_ + d];
      gin += u * W_ih[(2 * D_ + e) * D_ + d]; ghn += h * W_hh[(2 * D_ + e) * D_ + d];
    }
    float r = 1.f / (1.f + __expf(-(gir + ghr)));
    float z = 1.f / (1.f + __expf(-(giz + ghz)));
    float n = tanhf(gin + r * ghn);
    newh[i] = (1.f - z) * n + z * snl[s][e];
  }
  #pragma unroll
  for (int i = 0; i < 2; ++i) {
    int idx = threadIdx.x + i * 256;
    ns[idx >> 6][idx & 63] = newh[i];
  }
  __syncthreads();

  if (threadIdx.x < S_) {                     // LN stats for the MLP input
    float s = 0.f, s2 = 0.f;
    for (int d = 0; d < D_; ++d) { float v = ns[threadIdx.x][d]; s += v; s2 += v * v; }
    float mu = s * (1.f / D_);
    mus[threadIdx.x] = mu;
    rstds[threadIdx.x] = rsqrtf(s2 * (1.f / D_) - mu * mu + 1e-5f);
  }
  __syncthreads();
  #pragma unroll
  for (int i = 0; i < 2; ++i) {
    int idx = threadIdx.x + i * 256;
    int s = idx >> 6, d = idx & 63;
    lnn[s][d] = (ns[s][d] - mus[s]) * rstds[s] * gm[d] + bm[d];
  }
  __syncthreads();

  #pragma unroll
  for (int i = 0; i < 8; ++i) {               // h1 = relu(lnn @ W1^T + b1): 2048 outs
    int idx = threadIdx.x + i * 256;
    int s = idx >> 8, hh = idx & 255;
    float a1 = b1[hh];
    for (int d = 0; d < D_; ++d) a1 += lnn[s][d] * W1[hh * D_ + d];
    h1[s][hh] = fmaxf(a1, 0.f);
  }
  __syncthreads();

  #pragma unroll
  for (int i = 0; i < 2; ++i) {               // slots = ns + h1 @ W2^T + b2
    int idx = threadIdx.x + i * 256;
    int s = idx >> 6, e = idx & 63;
    float a2 = b2[e];
    for (int hh = 0; hh < H_; ++hh) a2 += h1[s][hh] * W2[e * H_ + hh];
    float v = ns[s][e] + a2;
    slots[(size_t)b * S_ * D_ + idx]     = v;
    slots_out[(size_t)b * S_ * D_ + idx] = v;   // final iteration's write wins
  }
}

// ---------------------------------------------------------------------------
extern "C" void kernel_launch(void* const* d_in, const int* in_sizes, int n_in,
                              void* d_out, int out_size, void* d_ws, size_t ws_size,
                              hipStream_t stream) {
  (void)in_sizes; (void)n_in; (void)out_size; (void)ws_size;
  const float* slots_in = (const float*)d_in[0];
  const float* features = (const float*)d_in[1];
  const float* gf   = (const float*)d_in[2];
  const float* bf   = (const float*)d_in[3];
  const float* Wk   = (const float*)d_in[4];
  const float* Wv   = (const float*)d_in[5];
  const float* Wq   = (const float*)d_in[6];
  const float* gs   = (const float*)d_in[7];
  const float* bs   = (const float*)d_in[8];
  const float* W_ih = (const float*)d_in[9];
  const float* W_hh = (const float*)d_in[10];
  const float* b_ih = (const float*)d_in[11];
  const float* b_hh = (const float*)d_in[12];
  const float* gm   = (const float*)d_in[13];
  const float* bm   = (const float*)d_in[14];
  const float* W1   = (const float*)d_in[15];
  const float* b1   = (const float*)d_in[16];
  const float* W2   = (const float*)d_in[17];
  const float* b2   = (const float*)d_in[18];

  char* wp = (char*)d_ws;
  auto carve = [&](size_t bytes) { void* r = wp; wp += (bytes + 255) & ~(size_t)255; return r; };
  _Float16* keys16 = (_Float16*)carve((size_t)B_ * F_ * D_ * 2);  // 32 MB
  _Float16* valT16 = (_Float16*)carve((size_t)B_ * D_ * F_ * 2);  // 32 MB
  _Float16* p16    = (_Float16*)carve((size_t)B_ * 16 * F_ * 2);  // 8 MB
  _Float16* q16    = (_Float16*)carve((size_t)B_ * 16 * D_ * 2);
  float*    sn     = (float*)carve((size_t)B_ * S_ * D_ * 4);
  float*    slots  = (float*)carve((size_t)B_ * S_ * D_ * 4);
  float*    invR   = (float*)carve((size_t)B_ * S_ * 4);
  _Float16* Wk16   = (_Float16*)carve((size_t)D_ * DIN_ * 2);
  _Float16* Wv16   = (_Float16*)carve((size_t)D_ * DIN_ * 2);

  float* out_slots = (float*)d_out;
  float* out_attn  = out_slots + (size_t)B_ * S_ * D_;

  hipMemcpyAsync(slots, slots_in, (size_t)B_ * S_ * D_ * sizeof(float),
                 hipMemcpyDeviceToDevice, stream);
  hipMemsetAsync(p16, 0, (size_t)B_ * 16 * F_ * 2, stream);  // rows 8..15 stay zero
  k_cvt_w<<<dim3((D_ * DIN_ + 255) / 256), 256, 0, stream>>>(Wk, Wv, Wk16, Wv16);
  k_ln_project<<<dim3(F_ / 128, B_), 128, 0, stream>>>(features, gf, bf, Wk16, Wv16,
                                                       keys16, valT16);
  for (int it = 0; it < 3; ++it) {
    k_slots_norm_q<<<B_, 256, 0, stream>>>(slots, gs, bs, Wq, sn, q16);
    k_attn<<<B_, 256, 0, stream>>>(q16, keys16, p16, out_attn, invR);
    k_update<<<B_, 256, 0, stream>>>(p16, valT16, invR, sn, W_ih, W_hh, b_ih, b_hh,
                                     gm, bm, W1, b1, W2, b2, slots, out_slots);
  }
}
